// PraxisAttention_30700426232522
// MI455X (gfx1250) — matrix operand order repaired and checked by reading, not compiled, not run on hardware
//
#include <hip/hip_runtime.h>
#include <hip/hip_bf16.h>

#define BB  2
#define SS  2048
#define DD  1024
#define HH  16
#define DHH 64

typedef __bf16 bf16_t;
typedef __attribute__((ext_vector_type(16))) __bf16 v16bf;
typedef __attribute__((ext_vector_type(8)))  __bf16 v8bf;
typedef __attribute__((ext_vector_type(8)))  float  v8f;

#define SHUF16(a, b) \
  __builtin_shufflevector(a, b, 0, 1, 2, 3, 4, 5, 6, 7, 8, 9, 10, 11, 12, 13, 14, 15)

// low 32 bits of a generic pointer to LDS == LDS byte address
__device__ __forceinline__ unsigned lds_lo32(const void* p) {
  return (unsigned)(uintptr_t)p;
}

// wave32 half-wave (16-lane) reductions: masks 1,2,4,8 stay within each half
__device__ __forceinline__ float rmax16(float v) {
#pragma unroll
  for (int m = 1; m < 16; m <<= 1) v = fmaxf(v, __shfl_xor(v, m, 32));
  return v;
}
__device__ __forceinline__ float rsum16(float v) {
#pragma unroll
  for (int m = 1; m < 16; m <<= 1) v += __shfl_xor(v, m, 32);
  return v;
}

__device__ __forceinline__ v8f wmma_bf16(const v16bf& a, const v16bf& b,
                                         const v8f& c) {
  return __builtin_amdgcn_wmma_f32_16x16x32_bf16(false, a, false, b, (short)0,
                                                 c, false, false);
}

// gfx1250 async copy global -> LDS (ASYNCcnt), GVS addressing
__device__ __forceinline__ void async_b128(unsigned lds_byte, unsigned voff_byte,
                                           const void* sbase) {
  asm volatile("global_load_async_to_lds_b128 %0, %1, %2"
               :: "v"(lds_byte), "v"(voff_byte), "s"(sbase)
               : "memory");
}
__device__ __forceinline__ void wait_async0() {
  asm volatile("s_wait_asynccnt 0x0" ::: "memory");
}

// ---------------------------------------------------------------------------
// f32 -> bf16 conversion
// ---------------------------------------------------------------------------
__global__ void cvt_f32_bf16(const float* __restrict__ in,
                             bf16_t* __restrict__ out, int n) {
  for (int i = blockIdx.x * blockDim.x + threadIdx.x; i < n;
       i += gridDim.x * blockDim.x)
    out[i] = (bf16_t)in[i];
}

// ---------------------------------------------------------------------------
// Tiled bf16 GEMM (MxK)@(KxN), f32 accum via v_wmma_f32_16x16x32_bf16.
// 256 thr (8 waves, 4x2), tile 128(M) x 64(N), K chunk 32.
//   A tile staged via global_load_async_to_lds_b128 (layout [m][k])
//   W tile staged transposed ([n][k]) so B-fragments are contiguous b128 reads
// out_mode 0: bf16 [B,H,S,DH] (QKV)   out_mode 1: f32 row-major (final)
// ---------------------------------------------------------------------------
__global__ __launch_bounds__(256) void gemm_bf16_wmma(
    const bf16_t* __restrict__ A, const bf16_t* __restrict__ W,
    void* __restrict__ outp, int M, int N, int K, int out_mode) {
  __shared__ __align__(32) bf16_t lA[128 * 32];   // [m][k]
  __shared__ __align__(32) bf16_t lWt[64 * 32];   // [n][k] (transposed)

  const int tid  = threadIdx.x;
  const int lane = tid & 31, wid = tid >> 5;
  const int lm = lane & 15, hi = lane >> 4;
  const int wm = wid & 3, wn = wid >> 2;  // 4(M) x 2(N) waves
  const int m0 = blockIdx.y * 128, n0 = blockIdx.x * 64;

  // ---- per-thread staging offsets (constant across K loop) ----
  // A: 4096 bf16 / 8-per-op = 512 ops -> 2 per thread
  unsigned aLds[2], aVoff[2];
#pragma unroll
  for (int t = 0; t < 2; ++t) {
    int idx = tid + t * 256;
    int row = idx >> 2, grp = idx & 3;                 // 4 b128 groups per row
    aLds[t]  = lds_lo32(lA) + (unsigned)((row * 32 + grp * 8) * 2);
    aVoff[t] = (unsigned)((((m0 + row) * K) + grp * 8) * 2);
  }
  // W transpose staging: 1 assignment per thread (k-pair x 4 n)
  const int kpair = tid >> 4, wgrp = tid & 15;
  const int k0r = kpair * 2;

  v8f acc[2][2];
#pragma unroll
  for (int i = 0; i < 2; ++i)
#pragma unroll
    for (int j = 0; j < 2; ++j)
#pragma unroll
      for (int r = 0; r < 8; ++r) acc[i][j][r] = 0.f;

  for (int kc = 0; kc < K; kc += 32) {
    // --- A tile: async global->LDS ---
    const bf16_t* aBase = A + kc;
#pragma unroll
    for (int t = 0; t < 2; ++t) async_b128(aLds[t], aVoff[t], aBase);

    // --- W tile: transposing stage, b32-packed stores ---
    {
      const unsigned short* wsrc =
          (const unsigned short*)W + (size_t)(kc + k0r) * N + n0 + wgrp * 4;
      ushort4 r0 = *(const ushort4*)wsrc;
      ushort4 r1 = *(const ushort4*)(wsrc + N);
      unsigned* dst = (unsigned*)lWt;
      dst[((wgrp * 4 + 0) * 32 + k0r) >> 1] = (unsigned)r0.x | ((unsigned)r1.x << 16);
      dst[((wgrp * 4 + 1) * 32 + k0r) >> 1] = (unsigned)r0.y | ((unsigned)r1.y << 16);
      dst[((wgrp * 4 + 2) * 32 + k0r) >> 1] = (unsigned)r0.z | ((unsigned)r1.z << 16);
      dst[((wgrp * 4 + 3) * 32 + k0r) >> 1] = (unsigned)r0.w | ((unsigned)r1.w << 16);
    }
    if (kc + 32 < K) {  // gfx1250 global_prefetch_b8 for next K slab
      __builtin_prefetch(&A[(size_t)(m0 + (tid >> 1)) * K + kc + 32], 0, 0);
      __builtin_prefetch(&W[(size_t)(kc + 32 + (tid >> 3)) * N + n0], 0, 0);
    }
    wait_async0();
    __syncthreads();

    // fragments: all contiguous b128 LDS reads
    v16bf aF[2], bF[2];
#pragma unroll
    for (int i = 0; i < 2; ++i) {
      const int mrow = wm * 32 + i * 16 + lm;
      v8bf x = *(const v8bf*)&lA[mrow * 32 + hi * 16];
      v8bf y = *(const v8bf*)&lA[mrow * 32 + hi * 16 + 8];
      aF[i] = SHUF16(x, y);
    }
#pragma unroll
    for (int j = 0; j < 2; ++j) {
      const int ncol = wn * 32 + j * 16 + lm;
      bF[j] = *(const v16bf*)&lWt[ncol * 32 + hi * 16];
    }
#pragma unroll
    for (int i = 0; i < 2; ++i)
#pragma unroll
      for (int j = 0; j < 2; ++j) acc[i][j] = wmma_bf16(aF[i], bF[j], acc[i][j]);
    __syncthreads();
  }

  // C layout: VGPR r -> row r + hi*8, col = lane&15
#pragma unroll
  for (int i = 0; i < 2; ++i)
#pragma unroll
    for (int j = 0; j < 2; ++j)
#pragma unroll
      for (int r = 0; r < 8; ++r) {
        int m = m0 + wm * 32 + i * 16 + r + hi * 8;
        int n = n0 + wn * 32 + j * 16 + lm;
        float val = acc[i][j][r];
        if (out_mode == 0) {
          int b = m >> 11, s = m & (SS - 1);
          int h = n >> 6, dh = n & (DHH - 1);
          ((bf16_t*)outp)[(((size_t)(b * HH + h) * SS) + s) * DHH + dh] =
              (bf16_t)val;
        } else {
          ((float*)outp)[(size_t)m * N + n] = val;
        }
      }
}

// ---------------------------------------------------------------------------
// Differential flash attention. Grid (S/64, H, B), 128 threads = 4 waves.
// K0/K1 chunks staged via async-to-LDS ([row][dh], matches score-B layout),
// V staged transposed ([dh][row]) so PV B-fragments are contiguous b128 reads.
// ---------------------------------------------------------------------------
#define STREAM_UPDATE(sT, MX, SM, OA)                                        \
  do {                                                                       \
    _Pragma("unroll") for (int r = 0; r < 8; ++r) {                          \
      float nm = rmax16(fmaxf(sT[0][r], sT[1][r]));                          \
      float mnew = fmaxf(MX[r], nm);                                         \
      float corr = __expf(MX[r] - mnew);                                     \
      MX[r] = mnew;                                                          \
      float p0 = __expf(sT[0][r] - mnew);                                    \
      float p1 = __expf(sT[1][r] - mnew);                                    \
      sT[0][r] = p0;                                                         \
      sT[1][r] = p1;                                                         \
      SM[r] = SM[r] * corr + rsum16(p0 + p1);                                \
      _Pragma("unroll") for (int j = 0; j < 4; ++j) OA[j][r] *= corr;        \
    }                                                                        \
    /* P (C layout) -> LDS -> A layout; wave-local, DS in-order */           \
    _Pragma("unroll") for (int t = 0; t < 2; ++t)                            \
        _Pragma("unroll") for (int r = 0; r < 8; ++r)                        \
            lP[wave][(r + hi * 8) * 32 + t * 16 + lm] = (bf16_t)sT[t][r];    \
    asm volatile("s_wait_dscnt 0x0" ::: "memory");                           \
    v16bf ap;                                                                \
    {                                                                        \
      v8bf x = *(const v8bf*)&lP[wave][lm * 32 + hi * 16];                   \
      v8bf y = *(const v8bf*)&lP[wave][lm * 32 + hi * 16 + 8];               \
      ap = SHUF16(x, y);                                                     \
    }                                                                        \
    _Pragma("unroll") for (int j = 0; j < 4; ++j)                            \
        OA[j] = wmma_bf16(ap, bv[j], OA[j]);                                 \
  } while (0)

__global__ __launch_bounds__(128) void diff_attn_kernel(
    const bf16_t* __restrict__ q0g, const bf16_t* __restrict__ q1g,
    const bf16_t* __restrict__ k0g, const bf16_t* __restrict__ k1g,
    const bf16_t* __restrict__ vg, const float* __restrict__ l0v,
    const float* __restrict__ l1v, const float* __restrict__ l2v,
    const float* __restrict__ l3v, const float* __restrict__ lnw,
    const float* __restrict__ lnb, bf16_t* __restrict__ attnout) {
  __shared__ __align__(32) bf16_t lK0[32 * 64];   // [row][dh]
  __shared__ __align__(32) bf16_t lK1[32 * 64];   // [row][dh]
  __shared__ __align__(32) bf16_t lVt[64 * 32];   // [dh][row] (transposed)
  __shared__ __align__(32) bf16_t lP[4][16 * 32];

  const int tid = threadIdx.x;
  const int lane = tid & 31, wave = tid >> 5;
  const int lm = lane & 15, hi = lane >> 4;
  const int b = blockIdx.z, h = blockIdx.y;
  const int qblk = blockIdx.x * 64;
  const int qw = qblk + wave * 16;

  const size_t headoff = (size_t)(b * HH + h) * SS * DHH;
  const bf16_t* q0p = q0g + headoff + (size_t)qw * DHH;
  const bf16_t* q1p = q1g + headoff + (size_t)qw * DHH;

  // Q fragments (A 16x32, two DH chunks, both streams): contiguous b128 reads
  v16bf aQ0[2], aQ1[2];
#pragma unroll
  for (int c2 = 0; c2 < 2; ++c2) {
    const int base = lm * DHH + c2 * 32 + hi * 16;
    {
      v8bf x = *(const v8bf*)&q0p[base];
      v8bf y = *(const v8bf*)&q0p[base + 8];
      aQ0[c2] = SHUF16(x, y);
    }
    {
      v8bf x = *(const v8bf*)&q1p[base];
      v8bf y = *(const v8bf*)&q1p[base + 8];
      aQ1[c2] = SHUF16(x, y);
    }
  }

  // staging offsets (constant across chunks): K async, 2048 bf16 -> 2 ops/thr
  unsigned kLds0[2], kLds1[2], kVoff[2];
#pragma unroll
  for (int t = 0; t < 2; ++t) {
    int idx = tid + t * 128;
    int row = idx >> 3, grp = idx & 7;                 // 8 b128 groups per row
    unsigned eo = (unsigned)((row * 64 + grp * 8) * 2);
    kVoff[t] = eo;
    kLds0[t] = lds_lo32(lK0) + eo;
    kLds1[t] = lds_lo32(lK1) + eo;
  }
  // V transposing stage: 256 assignments -> 2 per thread
  const int vsel0 = tid, vsel1 = tid + 128;

  v8f o0a[4], o1a[4];
  float mx0[8], sm0[8], mx1[8], sm1[8];
#pragma unroll
  for (int j = 0; j < 4; ++j)
#pragma unroll
    for (int r = 0; r < 8; ++r) {
      o0a[j][r] = 0.f;
      o1a[j][r] = 0.f;
    }
#pragma unroll
  for (int r = 0; r < 8; ++r) {
    mx0[r] = mx1[r] = -3.0e38f;
    sm0[r] = sm1[r] = 0.f;
  }

  const float scale = 0.125f;                         // rsqrt(DH)
  const float slope = exp2f(-0.5f * (float)(h + 1));  // 2^(-8(h+1)/H)
  const int nch = (qblk + 64) >> 5;                   // causal chunk count

  for (int c = 0; c < nch; ++c) {
    const int kb = c * 32;
    const bf16_t* sk0 = k0g + headoff + (size_t)kb * DHH;
    const bf16_t* sk1 = k1g + headoff + (size_t)kb * DHH;
    // K0/K1: async direct-to-LDS
#pragma unroll
    for (int t = 0; t < 2; ++t) {
      async_b128(kLds0[t], kVoff[t], sk0);
      async_b128(kLds1[t], kVoff[t], sk1);
    }
    // V: transpose into lVt with b32-packed stores
    {
      const unsigned short* vs =
          (const unsigned short*)(vg + headoff) + (size_t)kb * DHH;
      unsigned* dst = (unsigned*)lVt;
#pragma unroll
      for (int t = 0; t < 2; ++t) {
        int idx = (t == 0) ? vsel0 : vsel1;
        int rp = idx >> 4, grp = idx & 15;             // row-pair, 4-dh group
        int r0 = rp * 2;
        ushort4 a = *(const ushort4*)(vs + r0 * 64 + grp * 4);
        ushort4 bq = *(const ushort4*)(vs + (r0 + 1) * 64 + grp * 4);
        dst[((grp * 4 + 0) * 32 + r0) >> 1] = (unsigned)a.x | ((unsigned)bq.x << 16);
        dst[((grp * 4 + 1) * 32 + r0) >> 1] = (unsigned)a.y | ((unsigned)bq.y << 16);
        dst[((grp * 4 + 2) * 32 + r0) >> 1] = (unsigned)a.z | ((unsigned)bq.z << 16);
        dst[((grp * 4 + 3) * 32 + r0) >> 1] = (unsigned)a.w | ((unsigned)bq.w << 16);
      }
    }
    wait_async0();
    __syncthreads();

    if (kb <= qw + 15) {  // chunk intersects this wave's causal window
      v8f s0t[2], s1t[2];
#pragma unroll
      for (int t = 0; t < 2; ++t) {
        v8f a0, a1;
#pragma unroll
        for (int r = 0; r < 8; ++r) { a0[r] = 0.f; a1[r] = 0.f; }
#pragma unroll
        for (int d = 0; d < 2; ++d) {
          // B 32x16: contiguous 16 bf16 per lane ([row][dh] layout)
          v16bf bk0 = *(const v16bf*)&lK0[(t * 16 + lm) * 64 + d * 32 + hi * 16];
          v16bf bk1 = *(const v16bf*)&lK1[(t * 16 + lm) * 64 + d * 32 + hi * 16];
          a0 = wmma_bf16(aQ0[d], bk0, a0);
          a1 = wmma_bf16(aQ1[d], bk1, a1);
        }
        s0t[t] = a0;
        s1t[t] = a1;
      }
      // scale + ALiBi + causal mask
#pragma unroll
      for (int t = 0; t < 2; ++t)
#pragma unroll
        for (int r = 0; r < 8; ++r) {
          int row = qw + r + hi * 8;
          int col = kb + t * 16 + lm;
          float pd = (float)(row - col);
          float x0 = s0t[t][r] * scale - slope * pd;
          float x1 = s1t[t][r] * scale - slope * pd;
          s0t[t][r] = (col > row) ? -3.0e38f : x0;
          s1t[t][r] = (col > row) ? -3.0e38f : x1;
        }
      // V B-fragments: contiguous b128 reads from transposed tile
      v16bf bv[4];
#pragma unroll
      for (int j = 0; j < 4; ++j)
        bv[j] = *(const v16bf*)&lVt[(j * 16 + lm) * 32 + hi * 16];

      STREAM_UPDATE(s0t, mx0, sm0, o0a);
      STREAM_UPDATE(s1t, mx1, sm1, o1a);
    }
    __syncthreads();
  }

  // lambda = (0.8 - exp(l0.l1)) + (0.8 - exp(l2.l3))
  float lamb;
  {
    float d01 = 0.f, d23 = 0.f;
#pragma unroll
    for (int i = 0; i < DHH; ++i) {
      d01 += l0v[i] * l1v[i];
      d23 += l2v[i] * l3v[i];
    }
    lamb = (0.8f - __expf(d01)) + (0.8f - __expf(d23));
  }

  float vals[4][8];
#pragma unroll
  for (int j = 0; j < 4; ++j)
#pragma unroll
    for (int r = 0; r < 8; ++r)
      vals[j][r] = o0a[j][r] / sm0[r] - lamb * (o1a[j][r] / sm1[r]);

  // per-head LayerNorm over DH=64 (row spread across 16 lanes x 4 tiles)
#pragma unroll
  for (int r = 0; r < 8; ++r) {
    float mu =
        rsum16(vals[0][r] + vals[1][r] + vals[2][r] + vals[3][r]) * (1.f / 64.f);
    float var = 0.f;
#pragma unroll
    for (int j = 0; j < 4; ++j) {
      float d = vals[j][r] - mu;
      var += d * d;
    }
    var = rsum16(var) * (1.f / 64.f);
    float rstd = rsqrtf(var + 1e-5f);
    int row = qw + r + hi * 8;
#pragma unroll
    for (int j = 0; j < 4; ++j) {
      int dh = j * 16 + lm;
      float nv = (vals[j][r] - mu) * rstd * lnw[h * DHH + dh] + lnb[h * DHH + dh];
      attnout[((size_t)(b * SS + row)) * DD + h * DHH + dh] = (bf16_t)nv;
    }
  }
}

// ---------------------------------------------------------------------------
// Host orchestration
// ---------------------------------------------------------------------------
extern "C" void kernel_launch(void* const* d_in, const int* in_sizes, int n_in,
                              void* d_out, int out_size, void* d_ws,
                              size_t ws_size, hipStream_t stream) {
  const float* x = (const float*)d_in[0];
  const float* wsrc[6] = {(const float*)d_in[1], (const float*)d_in[2],
                          (const float*)d_in[3], (const float*)d_in[4],
                          (const float*)d_in[5], (const float*)d_in[6]};
  const float* l0  = (const float*)d_in[7];
  const float* l1  = (const float*)d_in[8];
  const float* l2  = (const float*)d_in[9];
  const float* l3  = (const float*)d_in[10];
  const float* lnw = (const float*)d_in[11];
  const float* lnb = (const float*)d_in[12];

  bf16_t* ws = (bf16_t*)d_ws;
  const size_t NX = (size_t)BB * SS * DD;        // 4,194,304
  const size_t NW = (size_t)DD * DD;             // 1,048,576
  const size_t NQ = (size_t)BB * HH * SS * DHH;  // 4,194,304

  bf16_t* Xb = ws;
  bf16_t* Wb[6];
  for (int i = 0; i < 6; ++i) Wb[i] = ws + NX + (size_t)i * NW;
  bf16_t* qkv[5];
  const size_t qbase = NX + 6 * NW;
  for (int i = 0; i < 5; ++i) qkv[i] = ws + qbase + (size_t)i * NQ;
  bf16_t* attnb = ws + qbase + 5 * NQ;

  // f32 -> bf16 staging
  cvt_f32_bf16<<<2048, 256, 0, stream>>>(x, Xb, (int)NX);
  for (int i = 0; i < 6; ++i)
    cvt_f32_bf16<<<1024, 256, 0, stream>>>(wsrc[i], Wb[i], (int)NW);

  // q0,q1,k0,k1,v projections
  dim3 ggrid(DD / 64, (BB * SS) / 128);  // (16, 32)
  for (int i = 0; i < 5; ++i)
    gemm_bf16_wmma<<<ggrid, 256, 0, stream>>>(Xb, Wb[i], qkv[i], BB * SS, DD,
                                              DD, 0);

  // differential flash attention + per-head LayerNorm
  diff_attn_kernel<<<dim3(SS / 64, HH, BB), 128, 0, stream>>>(
      qkv[0], qkv[1], qkv[2], qkv[3], qkv[4], l0, l1, l2, l3, lnw, lnb, attnb);

  // output projection -> f32 d_out
  gemm_bf16_wmma<<<ggrid, 256, 0, stream>>>(attnb, Wb[5], d_out, BB * SS, DD,
                                            DD, 1);
}
